// KoopmanOperator_1623497638653
// MI455X (gfx1250) — matrix operand order repaired and checked by reading, not compiled, not run on hardware
//
#include <hip/hip_runtime.h>

typedef __attribute__((ext_vector_type(16))) _Float16 v16h;
typedef __attribute__((ext_vector_type(8)))  _Float16 v8h;
typedef __attribute__((ext_vector_type(8)))  float    v8f;

#define TILE_WAVES 4
#define TSTEPS     256
#define DDIM       64
#define ROWH       72      // padded LDS row stride in halves (144B: 16B-aligned, bank-spread)
#define DT_STEP    0.01f

// DPP8 selectors: lane i reads sel[i] within its group of 8 (pairs never cross groups)
#define DPP8_SEL(a,b,c,d,e,f,g,h) \
    (((a)<<0)|((b)<<3)|((c)<<6)|((d)<<9)|((e)<<12)|((f)<<15)|((g)<<18)|((h)<<21))
#define SEL_XOR1 DPP8_SEL(1,0,3,2,5,4,7,6)   // swap within pair
#define SEL_EVEN DPP8_SEL(0,0,2,2,4,4,6,6)   // broadcast even lane of pair
#define SEL_ODD  DPP8_SEL(1,1,3,3,5,5,7,7)   // broadcast odd lane of pair

#if __has_builtin(__builtin_amdgcn_mov_dpp8)
#define HAVE_DPP8 1
__device__ __forceinline__ float dpp8_f(float v, int sel_xor1, int which) {
    int i = __builtin_bit_cast(int, v);
    if (which == 0)      i = __builtin_amdgcn_mov_dpp8(i, SEL_XOR1);
    else if (which == 1) i = __builtin_amdgcn_mov_dpp8(i, SEL_EVEN);
    else                 i = __builtin_amdgcn_mov_dpp8(i, SEL_ODD);
    (void)sel_xor1;
    return __builtin_bit_cast(float, i);
}
#else
#define HAVE_DPP8 0
#endif

__device__ __forceinline__ float fast_tanh(float x) {
#if __has_builtin(__builtin_amdgcn_tanhf)
    return __builtin_amdgcn_tanhf(x);
#else
    float t = __expf(2.0f * x);
    return (t - 1.0f) / (t + 1.0f);
#endif
}

__global__ __launch_bounds__(TILE_WAVES * 32, 1)
void koopman_scan_kernel(const float* __restrict__ x,
                         const float* __restrict__ W1,
                         const float* __restrict__ b1,
                         const float* __restrict__ W2,
                         const float* __restrict__ b2,
                         float* __restrict__ out)
{
    __shared__ __align__(32) _Float16 sW1[DDIM * DDIM];
    __shared__ __align__(32) _Float16 sW2[DDIM * DDIM];
    __shared__ __align__(32) _Float16 sYb[16 * ROWH];   // y staging (f16, row-major padded)
    __shared__ __align__(32) _Float16 sHb[16 * ROWH];   // h staging

    const int tid  = threadIdx.x;
    const int lane = tid & 31;
    const int wv   = tid >> 5;              // 0..3 : owns N-chunk nc = wv
    const int lo16 = lane & 15;
    const int hi   = (lane & 16) ? 1 : 0;
    const int Nown = lo16 + 16 * wv;        // this wave's output column

    // ---- one-time: f32 weights -> f16 in LDS (row-major, same index as global) ----
    for (int i = tid; i < DDIM * DDIM; i += blockDim.x) {
        sW1[i] = (_Float16)W1[i];
        sW2[i] = (_Float16)W2[i];
    }
    __syncthreads();

    // ---- resident B fragments for this wave's N-chunk only ----
    // B[k][n] = W[n][k]; B-frag lane holds column Nown, K = kc*32 + hi*8 + (0..7), then +16
    v16h w1f[2], w2f[2];
#pragma unroll
    for (int kc = 0; kc < 2; ++kc) {
        const int kb = kc * 32 + hi * 8;
        const v8h w1l = *(const v8h*)&sW1[Nown * DDIM + kb];
        const v8h w1h = *(const v8h*)&sW1[Nown * DDIM + kb + 16];
        const v8h w2l = *(const v8h*)&sW2[Nown * DDIM + kb];
        const v8h w2h = *(const v8h*)&sW2[Nown * DDIM + kb + 16];
        w1f[kc] = __builtin_shufflevector(w1l, w1h, 0,1,2,3,4,5,6,7,8,9,10,11,12,13,14,15);
        w2f[kc] = __builtin_shufflevector(w2l, w2h, 0,1,2,3,4,5,6,7,8,9,10,11,12,13,14,15);
    }
    const float b1v = b1[Nown];
    const float b2v = b2[Nown];

    const int rowbase = blockIdx.x * 16;    // one 16-row tile per block

    // ---- init y = x[:, 0, :] into this wave's C/D fragment slice ----
    // D-frag: VGPR r -> M = r + 8*hi ; N = Nown
    v8f yd;
#pragma unroll
    for (int r = 0; r < 8; ++r) {
        const size_t row = (size_t)(rowbase + r + 8 * hi);
        yd[r] = x[row * (TSTEPS * DDIM) + (size_t)Nown];
    }

    const bool  evenlane = (lane & 1) == 0;
    const float sgn      = evenlane ? 1.0f : -1.0f;   // n_odd = -(c*y_s - s*y_o)*e

#pragma unroll 1
    for (int t = 0; t < TSTEPS; ++t) {
        // ---- stage own y slice (f32 D-frag -> f16 tile rows) ----
#pragma unroll
        for (int r = 0; r < 8; ++r)
            sYb[(r + 8 * hi) * ROWH + Nown] = (_Float16)yd[r];
        __syncthreads();   // bar1: y visible; also fences last step's h reads

        // ---- full-K A fragments of y: lane row M=lo16 ----
        v16h a0, a1;
#pragma unroll
        for (int kc = 0; kc < 2; ++kc) {
            const v8h l8 = *(const v8h*)&sYb[lo16 * ROWH + kc * 32 + hi * 8];
            const v8h h8 = *(const v8h*)&sYb[lo16 * ROWH + kc * 32 + 16 + hi * 8];
            v16h a;
#pragma unroll
            for (int i = 0; i < 8; ++i) { a[i] = l8[i]; a[i + 8] = h8[i]; }
            if (kc == 0) a0 = a; else a1 = a;
        }

        // ---- h[:, Nown] = tanh(y @ W1^T + b1) ----
        v8f c;
#pragma unroll
        for (int i = 0; i < 8; ++i) c[i] = b1v;
        c = __builtin_amdgcn_wmma_f32_16x16x32_f16(false, a0, false, w1f[0],
                                                   (short)0, c, false, false);
        c = __builtin_amdgcn_wmma_f32_16x16x32_f16(false, a1, false, w1f[1],
                                                   (short)0, c, false, false);
        v8f hd;
#pragma unroll
        for (int i = 0; i < 8; ++i) hd[i] = fast_tanh(c[i]);

        // ---- stage own h slice ----
#pragma unroll
        for (int r = 0; r < 8; ++r)
            sHb[(r + 8 * hi) * ROWH + Nown] = (_Float16)hd[r];
        __syncthreads();   // bar2: h visible; also fences this step's y reads

        v16h ha0, ha1;
#pragma unroll
        for (int kc = 0; kc < 2; ++kc) {
            const v8h l8 = *(const v8h*)&sHb[lo16 * ROWH + kc * 32 + hi * 8];
            const v8h h8 = *(const v8h*)&sHb[lo16 * ROWH + kc * 32 + 16 + hi * 8];
            v16h a;
#pragma unroll
            for (int i = 0; i < 8; ++i) { a[i] = l8[i]; a[i + 8] = h8[i]; }
            if (kc == 0) ha0 = a; else ha1 = a;
        }

        // ---- p[:, Nown] = h @ W2^T + b2 ----
        v8f p;
#pragma unroll
        for (int i = 0; i < 8; ++i) p[i] = b2v;
        p = __builtin_amdgcn_wmma_f32_16x16x32_f16(false, ha0, false, w2f[0],
                                                   (short)0, p, false, false);
        p = __builtin_amdgcn_wmma_f32_16x16x32_f16(false, ha1, false, w2f[1],
                                                   (short)0, p, false, false);

        // ---- rotation update: (mu, omega) pairs live in adjacent lanes ----
        // mu = even-lane p, omega = odd-lane p (both lanes of the pair use them);
        // n_even = e*(c*y_s - s*y_o), n_odd = e*(s*y_o - c*y_s) = sgn * same form.
#pragma unroll
        for (int r = 0; r < 8; ++r) {
            float y_s = yd[r];
#if HAVE_DPP8
            float mu  = dpp8_f(p[r], 0, 1);   // SEL_EVEN
            float om  = dpp8_f(p[r], 0, 2);   // SEL_ODD
            float y_o = dpp8_f(y_s, 0, 0);    // SEL_XOR1
#else
            float p_o = __shfl_xor(p[r], 1, 32);
            float mu  = evenlane ? p[r] : p_o;
            float om  = evenlane ? p_o : p[r];
            float y_o = __shfl_xor(y_s, 1, 32);
#endif
            float e  = __expf(DT_STEP * mu);
            float cv = __cosf(DT_STEP * om);
            float sv = __sinf(DT_STEP * om);

            float nv = (cv * y_s - sv * y_o) * (e * sgn);
            yd[r] = nv;

            const size_t row = (size_t)(rowbase + r + 8 * hi);
            out[row * (TSTEPS * DDIM) + (size_t)t * DDIM + (size_t)Nown] = nv;
        }
    }
}

extern "C" void kernel_launch(void* const* d_in, const int* in_sizes, int n_in,
                              void* d_out, int out_size, void* d_ws, size_t ws_size,
                              hipStream_t stream) {
    const float* x  = (const float*)d_in[0];
    const float* W1 = (const float*)d_in[1];
    const float* b1 = (const float*)d_in[2];
    const float* W2 = (const float*)d_in[3];
    const float* b2 = (const float*)d_in[4];
    float* out = (float*)d_out;

    const int B = 4096;
    dim3 grid(B / 16);               // 256 blocks, one 16-row tile each
    dim3 block(TILE_WAVES * 32);     // 4 waves cooperating on the tile
    koopman_scan_kernel<<<grid, block, 0, stream>>>(x, W1, b1, W2, b2, out);
}
